// CrossAttention_gate_13640816132644
// MI455X (gfx1250) — compile-verified, matrix operand order checked by reading, not run on hardware
//
#include <hip/hip_runtime.h>

typedef __bf16 bf16_t;
typedef __attribute__((ext_vector_type(16))) __bf16 v16bf;
typedef __attribute__((ext_vector_type(8)))  __bf16 bf16x8;
typedef __attribute__((ext_vector_type(8)))  float  v8f;
typedef int v4i __attribute__((vector_size(16)));

#define GBM 256
#define GBN 64
#define GBK 32
#define GLDK 40   // 40 bf16 = 80B row stride: 16B-aligned, padded vs 32

#if defined(__gfx1250__) && __has_builtin(__builtin_amdgcn_global_load_async_to_lds_b128)
#define USE_ASYNC_LDS 1
#else
#define USE_ASYNC_LDS 0
#endif

#define AS1 __attribute__((address_space(1)))
#define AS3 __attribute__((address_space(3)))

__device__ __forceinline__ void wait_async0() {
#if USE_ASYNC_LDS
#if __has_builtin(__builtin_amdgcn_s_wait_asynccnt)
    __builtin_amdgcn_s_wait_asynccnt(0);
#else
    asm volatile("s_wait_asynccnt 0x0" ::: "memory");
#endif
#endif
}

#if USE_ASYNC_LDS
__device__ __forceinline__ void async_cp16(const bf16_t* g, bf16_t* l) {
    __builtin_amdgcn_global_load_async_to_lds_b128(
        (AS1 v4i*)(v4i*)(void*)g,
        (AS3 v4i*)(v4i*)(void*)l,
        0, 0);
}
#endif

__device__ __forceinline__ bf16_t f2bf(float f) {
    union { float f; unsigned u; } a; a.f = f;
    unsigned u = a.u;
    u += 0x7FFFu + ((u >> 16) & 1u);              // round-to-nearest-even
    union { unsigned short s; bf16_t b; } o;
    o.s = (unsigned short)(u >> 16);
    return o.b;
}

// ---------------------------------------------------------------------------
// Generic bf16 WMMA GEMM:  C[m,n] = sum_k A[m,k]*B[n,k]   (K-contiguous A & B)
// A elem (m,k) at A[m*sA + k]; B elem (n,k) at B[n*sB + k]
// epilogue: v = (acc + bias[n]) * alpha; optional relu; store f32 or bf16 at
// D[m*sDm + n*sDn].
// Block tile 256x64, 8 waves = 8 m-waves, wave tile 32x64 -> 8 WMMA/wave/step.
// Double-buffered LDS (one barrier per K-step); async global->LDS when
// available (ASYNCcnt path), sync global_load+ds_store fallback otherwise.
// ---------------------------------------------------------------------------
__global__ __launch_bounds__(256) void gemm_bf16_wmma(
    const bf16_t* __restrict__ A, long sA,
    const bf16_t* __restrict__ B, long sB,
    int M, int Nout, int K,
    const float* __restrict__ bias, float alpha, int relu,
    float* __restrict__ Df32, bf16_t* __restrict__ Dbf,
    long sDm, long sDn)
{
    __shared__ bf16_t lA[2][GBM * GLDK];
    __shared__ bf16_t lB[2][GBN * GLDK];

    const int tid  = threadIdx.x;
    const int lane = tid & 31;
    const int wave = tid >> 5;
    const int wm   = wave * 32;         // wave m-origin in tile (8 waves * 32)
    const long blockM = (long)blockIdx.x * GBM;
    const long blockN = (long)blockIdx.y * GBN;

    v8f acc[2][4];
    const v8f vzero = {0.f,0.f,0.f,0.f,0.f,0.f,0.f,0.f};
    for (int i = 0; i < 2; ++i)
        for (int j = 0; j < 4; ++j) acc[i][j] = vzero;

    const int r0 = tid >> 2;          // 0..63
    const int kc = (tid & 3) * 8;     // 0,8,16,24

    const int frow = lane & 15;
    const int kh   = (lane >> 4) * 8; // K base for h[0] (h[1] at +16)

    union Frag { v16bf v; bf16x8 h[2]; };

    // ---- tile movement helpers -------------------------------------------
#if USE_ASYNC_LDS
    auto issueTile = [&](int k0, int bufIdx) {
        const bf16_t* a0 = A + (blockM + r0) * sA + k0 + kc;
        bf16_t* la = &lA[bufIdx][r0 * GLDK + kc];
        async_cp16(a0,             la);
        async_cp16(a0 +  64 * sA,  la +  64 * GLDK);
        async_cp16(a0 + 128 * sA,  la + 128 * GLDK);
        async_cp16(a0 + 192 * sA,  la + 192 * GLDK);
        async_cp16(B + (blockN + r0) * sB + k0 + kc,
                   &lB[bufIdx][r0 * GLDK + kc]);
    };
#else
    bf16x8 stg[5];
    auto loadTile = [&](int k0) {
        const bf16_t* a0 = A + (blockM + r0) * sA + k0 + kc;
        stg[0] = *(const bf16x8*)a0;
        stg[1] = *(const bf16x8*)(a0 +  64 * sA);
        stg[2] = *(const bf16x8*)(a0 + 128 * sA);
        stg[3] = *(const bf16x8*)(a0 + 192 * sA);
        stg[4] = *(const bf16x8*)(B + (blockN + r0) * sB + k0 + kc);
    };
    auto storeTile = [&](int bufIdx) {
        bf16_t* la = &lA[bufIdx][r0 * GLDK + kc];
        *(bf16x8*)(la)              = stg[0];
        *(bf16x8*)(la +  64 * GLDK) = stg[1];
        *(bf16x8*)(la + 128 * GLDK) = stg[2];
        *(bf16x8*)(la + 192 * GLDK) = stg[3];
        *(bf16x8*)&lB[bufIdx][r0 * GLDK + kc] = stg[4];
    };
#endif

    auto compute = [&](int bufIdx) {
        Frag fa[2], fb[4];
        for (int i = 0; i < 2; ++i) {
            int m = wm + i * 16 + frow;
            fa[i].h[0] = *(const bf16x8*)&lA[bufIdx][m * GLDK + kh];
            fa[i].h[1] = *(const bf16x8*)&lA[bufIdx][m * GLDK + kh + 16];
        }
        for (int j = 0; j < 4; ++j) {
            int n = j * 16 + frow;
            fb[j].h[0] = *(const bf16x8*)&lB[bufIdx][n * GLDK + kh];
            fb[j].h[1] = *(const bf16x8*)&lB[bufIdx][n * GLDK + kh + 16];
        }
        for (int i = 0; i < 2; ++i)
            for (int j = 0; j < 4; ++j)
                acc[i][j] = __builtin_amdgcn_wmma_f32_16x16x32_bf16(
                    false, fa[i].v, false, fb[j].v, (short)0, acc[i][j],
                    false, false);
    };

    // ---- main pipeline: double-buffered, one barrier per K-step ----------
    int p = 0;
#if USE_ASYNC_LDS
    issueTile(0, 0);
    for (int k0 = 0; k0 < K; k0 += GBK) {
        wait_async0();               // this wave's async LDS writes landed
        __syncthreads();             // everyone's; prev frag reads done
        if (k0 + GBK < K) issueTile(k0 + GBK, p ^ 1);
        compute(p);
        p ^= 1;
    }
#else
    loadTile(0);
    storeTile(0);
    for (int k0 = 0; k0 < K; k0 += GBK) {
        __syncthreads();             // buf[p] visible; prev reads of p^1 done
        bool more = (k0 + GBK < K);
        if (more) loadTile(k0 + GBK);
        compute(p);
        if (more) storeTile(p ^ 1);
        p ^= 1;
    }
#endif

    // epilogue per D lane layout: VGPR r -> M=r(+8), lane&15 -> N
    const int nLane  = lane & 15;
    const int mShift = (lane >> 4) * 8;
    for (int i = 0; i < 2; ++i)
        for (int j = 0; j < 4; ++j) {
            long nG = blockN + j * 16 + nLane;
            float bv = bias ? bias[nG] : 0.f;
            for (int r = 0; r < 8; ++r) {
                long mG = blockM + wm + i * 16 + r + mShift;
                float v = (acc[i][j][r] + bv) * alpha;
                if (relu) v = fmaxf(v, 0.f);
                if (Df32) Df32[mG * sDm + nG * sDn] = v;
                else      Dbf [mG * sDm + nG * sDn] = f2bf(v);
            }
        }
}

// ---------------------------------------------------------------------------
// Prep: masks + bf16 pixel-major (K-contiguous) tensors
// ---------------------------------------------------------------------------
__global__ __launch_bounds__(256) void prep_inputs(
    const float* __restrict__ x, const float* __restrict__ y,
    const float* __restrict__ s_m, const float* __restrict__ q_m,
    bf16_t* __restrict__ Xt, bf16_t* __restrict__ Xfgt,
    bf16_t* __restrict__ Yfgt, float* __restrict__ qbin, int C, int NP)
{
    long idx = (long)blockIdx.x * blockDim.x + threadIdx.x;
    long total = (long)C * NP;
    if (idx >= total) return;
    long c = idx / NP, n = idx % NP;
    float xv = x[idx], yv = y[idx];
    float sm = s_m[n];
    float qb = (q_m[n] > 0.8f) ? 1.f : 0.f;
    if (c == 0) qbin[n] = qb;
    Xt  [n * C + c] = f2bf(xv);
    Xfgt[n * C + c] = f2bf(xv * sm);
    Yfgt[n * C + c] = f2bf(yv * qb);
}

__global__ __launch_bounds__(256) void cvt_bf16(
    const float* __restrict__ s, bf16_t* __restrict__ d, long n)
{
    long i = (long)blockIdx.x * blockDim.x + threadIdx.x;
    if (i < n) d[i] = f2bf(s[i]);
}

// ---------------------------------------------------------------------------
// Reductions
// ---------------------------------------------------------------------------
__global__ __launch_bounds__(256) void reduce_sum_ext(
    const float* __restrict__ S, long total, int useMax,
    double* __restrict__ pSum, float* __restrict__ pExt)
{
    __shared__ double sd[256];
    __shared__ float  se[256];
    int tid = threadIdx.x;
    long stride = (long)gridDim.x * 256;
    double s = 0.0;
    float e = useMax ? -3.4e38f : 3.4e38f;
    for (long i = (long)blockIdx.x * 256 + tid; i < total; i += stride) {
        float v = S[i];
        s += (double)v;
        e = useMax ? fmaxf(e, v) : fminf(e, v);
    }
    sd[tid] = s; se[tid] = e; __syncthreads();
    for (int o = 128; o > 0; o >>= 1) {
        if (tid < o) {
            sd[tid] += sd[tid + o];
            se[tid] = useMax ? fmaxf(se[tid], se[tid + o])
                             : fminf(se[tid], se[tid + o]);
        }
        __syncthreads();
    }
    if (tid == 0) { pSum[blockIdx.x] = sd[0]; pExt[blockIdx.x] = se[0]; }
}

__global__ __launch_bounds__(256) void reduce_thr_final(
    const double* __restrict__ pSum, const float* __restrict__ pExt,
    int nPart, int useMax, double count, float* __restrict__ thrOut)
{
    __shared__ double sd[256];
    __shared__ float  se[256];
    int tid = threadIdx.x;
    double s = 0.0;
    float e = useMax ? -3.4e38f : 3.4e38f;
    for (int i = tid; i < nPart; i += 256) {
        s += pSum[i];
        e = useMax ? fmaxf(e, pExt[i]) : fminf(e, pExt[i]);
    }
    sd[tid] = s; se[tid] = e; __syncthreads();
    for (int o = 128; o > 0; o >>= 1) {
        if (tid < o) {
            sd[tid] += sd[tid + o];
            se[tid] = useMax ? fmaxf(se[tid], se[tid + o])
                             : fminf(se[tid], se[tid + o]);
        }
        __syncthreads();
    }
    if (tid == 0) thrOut[0] = (float)(0.5 * ((double)se[0] + sd[0] / count));
}

// row softmax with threshold mask; converts f32 scores -> bf16 attn IN PLACE
__global__ __launch_bounds__(256) void softmax_row(
    float* __restrict__ S, const float* __restrict__ thrP, int Ncols)
{
    __shared__ float red[256];
    const int tid = threadIdx.x;
    float* rp = S + (long)blockIdx.x * Ncols;
    bf16_t* op = (bf16_t*)rp;
    const int PER = 16;                 // Ncols == 4096
    float vals[PER];
    float thr = thrP[0];
    float m = -3.4e38f;
    for (int i = 0; i < PER; ++i) {     // all reads happen before first barrier
        float v = rp[tid * PER + i];
        v = (v < thr) ? -999.0f : v;
        vals[i] = v;
        m = fmaxf(m, v);
    }
    red[tid] = m; __syncthreads();
    for (int o = 128; o > 0; o >>= 1) {
        if (tid < o) red[tid] = fmaxf(red[tid], red[tid + o]);
        __syncthreads();
    }
    float rmax = red[0]; __syncthreads();
    float s = 0.f;
    for (int i = 0; i < PER; ++i) { vals[i] = __expf(vals[i] - rmax); s += vals[i]; }
    red[tid] = s; __syncthreads();
    for (int o = 128; o > 0; o >>= 1) {
        if (tid < o) red[tid] += red[tid + o];
        __syncthreads();
    }
    float rinv = 1.f / red[0];
    for (int i = 0; i < PER; ++i) op[tid * PER + i] = f2bf(vals[i] * rinv);
}

__global__ __launch_bounds__(256) void reduce_sumsq(
    const float* __restrict__ H, long total,
    double* __restrict__ pS, double* __restrict__ pQ)
{
    __shared__ double sd[256];
    __shared__ double sq[256];
    int tid = threadIdx.x;
    long stride = (long)gridDim.x * 256;
    double s = 0.0, q = 0.0;
    for (long i = (long)blockIdx.x * 256 + tid; i < total; i += stride) {
        double v = (double)H[i];
        s += v; q += v * v;
    }
    sd[tid] = s; sq[tid] = q; __syncthreads();
    for (int o = 128; o > 0; o >>= 1) {
        if (tid < o) { sd[tid] += sd[tid + o]; sq[tid] += sq[tid + o]; }
        __syncthreads();
    }
    if (tid == 0) { pS[blockIdx.x] = sd[0]; pQ[blockIdx.x] = sq[0]; }
}

__global__ __launch_bounds__(256) void stats_final(
    const double* __restrict__ pS, const double* __restrict__ pQ,
    int nPart, double count, float* __restrict__ stats)
{
    __shared__ double sd[256];
    __shared__ double sq[256];
    int tid = threadIdx.x;
    double s = 0.0, q = 0.0;
    for (int i = tid; i < nPart; i += 256) { s += pS[i]; q += pQ[i]; }
    sd[tid] = s; sq[tid] = q; __syncthreads();
    for (int o = 128; o > 0; o >>= 1) {
        if (tid < o) { sd[tid] += sd[tid + o]; sq[tid] += sq[tid + o]; }
        __syncthreads();
    }
    if (tid == 0) {
        double mu  = sd[0] / count;
        double var = sq[0] / count - mu * mu;
        stats[0] = (float)mu;
        stats[1] = (float)(1.0 / sqrt(var + 1e-5));
    }
}

// LayerNorm affine + mask gating + residual blend.  H2 stored [C, NP].
__global__ __launch_bounds__(256) void finalize_out(
    const float* __restrict__ H2, const float* __restrict__ stats,
    const float* __restrict__ gamma, const float* __restrict__ beta,
    const float* __restrict__ orig, const float* __restrict__ mask,
    float* __restrict__ out, int C, int NP)
{
    long idx = (long)blockIdx.x * blockDim.x + threadIdx.x;
    long total = (long)C * NP;
    if (idx >= total) return;
    long n = idx % NP;
    float mu = stats[0], rstd = stats[1];
    float v = (H2[idx] - mu) * rstd * gamma[idx] + beta[idx];
    float mk = mask[n];
    out[idx] = v * mk + orig[idx] * (1.f - mk);
}

// ---------------------------------------------------------------------------
// Host orchestration
// ---------------------------------------------------------------------------
static inline void gemm(const bf16_t* A, long sA, const bf16_t* B, long sB,
                        int M, int Nn, int K, const float* bias, float alpha,
                        int relu, float* Df, bf16_t* Db, long sDm, long sDn,
                        hipStream_t st)
{
    dim3 g(M / GBM, Nn / GBN), b(256);
    gemm_bf16_wmma<<<g, b, 0, st>>>(A, sA, B, sB, M, Nn, K, bias, alpha, relu,
                                    Df, Db, sDm, sDn);
}

static void run_branch(const bf16_t* qin, const bf16_t* kin, const bf16_t* vin,
                       int useMax,
                       const bf16_t* Wqb, const bf16_t* Wkb, const bf16_t* Wvb,
                       const bf16_t* W1b, const bf16_t* W2b,
                       const float* bq, const float* bk, const float* bvv,
                       const float* b1, const float* b2,
                       const float* gamma, const float* beta,
                       const float* orig, const float* mask, float* outp,
                       float* wS, bf16_t* wQ, bf16_t* wK, bf16_t* wV,
                       bf16_t* wO, bf16_t* wH1, float* wH2,
                       double* pS, float* pE, double* pQ2, float* scal,
                       hipStream_t st)
{
    const int C = 512, NP = 4096, D = 64;
    // projections (q scaled by d^-0.5 = 0.125 after bias, matching reference)
    gemm(qin, C, Wqb, C, NP, D, C, bq, 0.125f, 0, nullptr, wQ, D, 1, st);
    gemm(kin, C, Wkb, C, NP, D, C, bk, 1.f, 0, nullptr, wK, D, 1, st);
    gemm(vin, C, Wvb, C, NP, C, C, bvv, 1.f, 0, nullptr, wV, 1, NP, st); // [C,NP]
    // S = Q K^T  (f32 scores)
    gemm(wQ, D, wK, D, NP, NP, D, nullptr, 1.f, 0, wS, nullptr, NP, 1, st);
    // global threshold = 0.5*(extremum + mean)
    reduce_sum_ext<<<1024, 256, 0, st>>>(wS, (long)NP * NP, useMax, pS, pE);
    reduce_thr_final<<<1, 256, 0, st>>>(pS, pE, 1024, useMax,
                                        (double)NP * (double)NP, scal);
    // masked softmax, f32 -> bf16 in place (bf16 row stride = 2*NP elements)
    softmax_row<<<NP, 256, 0, st>>>(wS, scal, NP);
    // O = attn @ V^T   -> [NP, C] bf16
    gemm((const bf16_t*)wS, 2L * NP, wV, NP, NP, C, NP, nullptr, 1.f, 0,
         nullptr, wO, C, 1, st);
    // MLP
    gemm(wO, C, W1b, C, NP, C, C, b1, 1.f, 1, nullptr, wH1, C, 1, st);
    gemm(wH1, C, W2b, C, NP, C, C, b2, 1.f, 0, wH2, nullptr, 1, NP, st); // [C,NP]
    // LayerNorm stats over all C*NP elements
    reduce_sumsq<<<512, 256, 0, st>>>(wH2, (long)NP * C, pS, pQ2);
    stats_final<<<1, 256, 0, st>>>(pS, pQ2, 512, (double)NP * C, scal + 1);
    finalize_out<<<(C * NP + 255) / 256, 256, 0, st>>>(
        wH2, scal + 1, gamma, beta, orig, mask, outp, C, NP);
}

extern "C" void kernel_launch(void* const* d_in, const int* in_sizes, int n_in,
                              void* d_out, int out_size, void* d_ws,
                              size_t ws_size, hipStream_t stream)
{
    (void)in_sizes; (void)n_in; (void)out_size; (void)ws_size;
    const float* x    = (const float*)d_in[0];
    const float* y    = (const float*)d_in[1];
    const float* s_m  = (const float*)d_in[2];
    const float* q_m  = (const float*)d_in[3];
    const float* Wq   = (const float*)d_in[4];
    const float* bq   = (const float*)d_in[5];
    const float* Wk   = (const float*)d_in[6];
    const float* bk   = (const float*)d_in[7];
    const float* Wv   = (const float*)d_in[8];
    const float* bv   = (const float*)d_in[9];
    const float* W1   = (const float*)d_in[10];
    const float* b1   = (const float*)d_in[11];
    const float* W2   = (const float*)d_in[12];
    const float* b2   = (const float*)d_in[13];
    const float* gm   = (const float*)d_in[14];
    const float* bt   = (const float*)d_in[15];
    float* out = (float*)d_out;

    const int C = 512, NP = 4096, D = 64;

    // carve workspace (256B aligned)
    char* w = (char*)d_ws;
    size_t off = 0;
    auto carve = [&](size_t bytes) -> void* {
        void* p = w + off;
        off = (off + bytes + 255) & ~(size_t)255;
        return p;
    };
    float*  wS    = (float*)carve((size_t)NP * NP * 4);  // scores / attn(bf16 in place)
    bf16_t* wQ    = (bf16_t*)carve((size_t)NP * D * 2);
    bf16_t* wK    = (bf16_t*)carve((size_t)NP * D * 2);
    bf16_t* wV    = (bf16_t*)carve((size_t)C * NP * 2);
    bf16_t* wO    = (bf16_t*)carve((size_t)NP * C * 2);
    bf16_t* wH1   = (bf16_t*)carve((size_t)NP * C * 2);
    float*  wH2   = (float*)carve((size_t)NP * C * 4);
    bf16_t* wXt   = (bf16_t*)carve((size_t)NP * C * 2);
    bf16_t* wXfgt = (bf16_t*)carve((size_t)NP * C * 2);
    bf16_t* wYfgt = (bf16_t*)carve((size_t)NP * C * 2);
    bf16_t* wWqb  = (bf16_t*)carve((size_t)D * C * 2);
    bf16_t* wWkb  = (bf16_t*)carve((size_t)D * C * 2);
    bf16_t* wWvb  = (bf16_t*)carve((size_t)C * C * 2);
    bf16_t* wW1b  = (bf16_t*)carve((size_t)C * C * 2);
    bf16_t* wW2b  = (bf16_t*)carve((size_t)C * C * 2);
    float*  wQbin = (float*)carve((size_t)NP * 4);
    double* pS    = (double*)carve(1024 * 8);
    float*  pE    = (float*)carve(1024 * 4);
    double* pQ2   = (double*)carve(1024 * 8);
    float*  scal  = (float*)carve(256);   // [0]=thr, [1]=mu, [2]=rstd

    // input prep + weight conversion
    prep_inputs<<<(C * NP + 255) / 256, 256, 0, stream>>>(
        x, y, s_m, q_m, wXt, wXfgt, wYfgt, wQbin, C, NP);
    cvt_bf16<<<((long)D * C + 255) / 256, 256, 0, stream>>>(Wq, wWqb, (long)D * C);
    cvt_bf16<<<((long)D * C + 255) / 256, 256, 0, stream>>>(Wk, wWkb, (long)D * C);
    cvt_bf16<<<((long)C * C + 255) / 256, 256, 0, stream>>>(Wv, wWvb, (long)C * C);
    cvt_bf16<<<((long)C * C + 255) / 256, 256, 0, stream>>>(W1, wW1b, (long)C * C);
    cvt_bf16<<<((long)C * C + 255) / 256, 256, 0, stream>>>(W2, wW2b, (long)C * C);

    // branch 1: q from x_fg, k/v from y_fg, max-threshold, gate by s_m, blend x
    run_branch(wXfgt, wYfgt, wYfgt, 1, wWqb, wWkb, wWvb, wW1b, wW2b,
               bq, bk, bv, b1, b2, gm, bt, x, s_m, out,
               wS, wQ, wK, wV, wO, wH1, wH2, pS, pE, pQ2, scal, stream);
    // branch 2: q from y_fg, k/v from x, min-threshold, gate by q_bin, blend y
    run_branch(wYfgt, wXt, wXt, 0, wWqb, wWkb, wWvb, wW1b, wW2b,
               bq, bk, bv, b1, b2, gm, bt, y, wQbin, out + (size_t)C * NP,
               wS, wQ, wK, wV, wO, wH1, wH2, pS, pE, pQ2, scal, stream);
}
